// TransMatcher_44135083933763
// MI455X (gfx1250) — matrix operand order repaired and checked by reading, not compiled, not run on hardware
//
#include <hip/hip_runtime.h>

typedef __attribute__((ext_vector_type(16))) __bf16 v16bf;
typedef __attribute__((ext_vector_type(8)))  __bf16 v8bf;
typedef __attribute__((ext_vector_type(8)))  float  v8f;
typedef int v4i_t __attribute__((vector_size(16)));   // matches builtin proto

#define AS1 __attribute__((address_space(1)))
#define AS3 __attribute__((address_space(3)))

namespace {
constexpr int QN  = 48;
constexpr int KN  = 96;
constexpr int SEQ = 192;   // 24*8
constexpr int DD  = 512;
constexpr int DFF = 2048;
constexpr float EPSV = 1e-5f;

union BfVec16 { v16bf v; v8bf h[2]; __bf16 e[16]; };

__device__ __forceinline__ void split_bf16(float x, __bf16& hi, __bf16& lo) {
  __bf16 h = (__bf16)x;          // RNE to bf16
  hi = h;
  lo = (__bf16)(x - (float)h);   // residual, also bf16
}

// ---- CDNA5 async global->LDS copy (ASYNCcnt-tracked), 16B per lane ----
template <int OFS>
__device__ __forceinline__ void asyncLoad16(const __bf16* g, __bf16* lds) {
#if __has_builtin(__builtin_amdgcn_global_load_async_to_lds_b128)
  __builtin_amdgcn_global_load_async_to_lds_b128(
      (AS1 v4i_t*)g, (AS3 v4i_t*)lds, OFS, 0);
#else
  unsigned l = (unsigned)(uintptr_t)(AS3 void*)lds;
  asm volatile("global_load_async_to_lds_b128 %0, %1, off offset:%2"
               :: "v"(l), "v"((unsigned long long)(uintptr_t)g), "i"(OFS)
               : "memory");
#endif
}

template <int N>
__device__ __forceinline__ void waitAsync() {
#if __has_builtin(__builtin_amdgcn_s_wait_asynccnt)
  __builtin_amdgcn_s_wait_asynccnt(N);
#else
  asm volatile("s_wait_asynccnt %0" :: "i"(N) : "memory");
#endif
}
} // namespace

// ---------------------------------------------------------------------------
// Kernel 0: one-shot fp32 -> bf16 hi/lo plane conversion (done once per tensor
// instead of once per (q,k) block => kills the hot-loop VALU conversion cost).
// ---------------------------------------------------------------------------
__global__ __launch_bounds__(256)
void convert_kernel(const float* __restrict__ src,
                    __bf16* __restrict__ hi, __bf16* __restrict__ lo, int n8) {
  int i = blockIdx.x * 256 + threadIdx.x;   // 8 floats per thread
  if (i >= n8) return;
  const float4* s = (const float4*)src + (size_t)i * 2;
  float4 f0 = s[0], f1 = s[1];
  float tv[8] = {f0.x, f0.y, f0.z, f0.w, f1.x, f1.y, f1.z, f1.w};
  union { __bf16 e[8]; uint4 u; } ph, pl;
#pragma unroll
  for (int e = 0; e < 8; ++e) split_bf16(tv[e], ph.e[e], pl.e[e]);
  *(uint4*)(hi + (size_t)i * 8) = ph.u;
  *(uint4*)(lo + (size_t)i * 8) = pl.u;
}

// ---------------------------------------------------------------------------
// Kernel 1: fused batched GEMM (192x192x512 per (q,k), bf16x3 WMMA) + max-pool.
// One block per pair, 12 waves; wave w owns s-strip [16w,16w+16).
// A operand (features) streamed global->regs; B operand (memory) double-buffered
// into LDS via async global->LDS copies (ASYNCcnt).
// pooled row 2*pair   = max over s (indexed by t)
// pooled row 2*pair+1 = max over t (indexed by s)
// ---------------------------------------------------------------------------
__global__ __launch_bounds__(384)
void score_pool_kernel(const __bf16* __restrict__ memHi, const __bf16* __restrict__ memLo,
                       const __bf16* __restrict__ featHi, const __bf16* __restrict__ featLo,
                       float* __restrict__ pooled) {
  __shared__ __bf16 BufHi[2][SEQ * 32];   // 2 x 12 KB
  __shared__ __bf16 BufLo[2][SEQ * 32];   // 2 x 12 KB
  __shared__ float  colPart[12 * SEQ];    // 9 KB

  const int pair = blockIdx.x;
  const int q = pair / KN;
  const int k = pair - q * KN;
  const __bf16* __restrict__ aHiG = featHi + (size_t)k * SEQ * DD;  // WMMA A (M=s)
  const __bf16* __restrict__ aLoG = featLo + (size_t)k * SEQ * DD;
  const __bf16* __restrict__ bHiG = memHi  + (size_t)q * SEQ * DD;  // WMMA B (N=t)
  const __bf16* __restrict__ bLoG = memLo  + (size_t)q * SEQ * DD;

  const int tid  = threadIdx.x;
  const int lane = tid & 31;
  const int wave = tid >> 5;     // 0..11
  const int s0   = wave * 16;

  // async-stage assignment: thread -> (plane, row); 4 x b128 per thread/chunk
  const int prow = tid % SEQ;            // 0..191
  const int pl   = tid / SEQ;            // 0 = hi, 1 = lo
  const __bf16* srcRow = (pl ? bLoG : bHiG) + (size_t)prow * DD;

  const v8f vzero = {0.f, 0.f, 0.f, 0.f, 0.f, 0.f, 0.f, 0.f};
  v8f acc[12];
#pragma unroll
  for (int i = 0; i < 12; ++i) acc[i] = vzero;

  // prologue: stage chunk 0 into buffer 0
  {
    const __bf16* g = srcRow;            // d0 = 0
    __bf16* d = pl ? &BufLo[0][prow * 32] : &BufHi[0][prow * 32];
    asyncLoad16<0>(g, d);
    asyncLoad16<16>(g, d);
    asyncLoad16<32>(g, d);
    asyncLoad16<48>(g, d);
  }

  const int arow = s0 + (lane & 15);
  const int kb   = (lane < 16) ? 0 : 8;    // ISA 16-bit A 16x32 layout
  const int koff = (lane < 16) ? 0 : 8;    // ISA 16-bit B 32x16 layout

  for (int kk = 0; kk < 16; ++kk) {
    const int cur = kk & 1;
    if (kk < 15) {                         // prefetch next chunk into other buffer
      const __bf16* g = srcRow + (kk + 1) * 32;
      __bf16* d = pl ? &BufLo[cur ^ 1][prow * 32] : &BufHi[cur ^ 1][prow * 32];
      asyncLoad16<0>(g, d);
      asyncLoad16<16>(g, d);
      asyncLoad16<32>(g, d);
      asyncLoad16<48>(g, d);
      waitAsync<4>();                      // our 4 copies for chunk kk done
    } else {
      waitAsync<0>();
    }
    __syncthreads();                       // everyone's copies for chunk kk done

    // A operand: direct global b128 loads from precomputed planes
    const __bf16* pA = aHiG + (size_t)arow * DD + kk * 32 + kb;
    const __bf16* pAl = aLoG + (size_t)arow * DD + kk * 32 + kb;
    BfVec16 aH, aL;
    aH.h[0] = *(const v8bf*)pA;
    aH.h[1] = *(const v8bf*)(pA + 16);
    aL.h[0] = *(const v8bf*)pAl;
    aL.h[1] = *(const v8bf*)(pAl + 16);

    const __bf16* lH = &BufHi[cur][0];
    const __bf16* lL = &BufLo[cur][0];
#pragma unroll
    for (int tt = 0; tt < 12; ++tt) {
      const int trow = tt * 16 + (lane & 15);
      BfVec16 bH, bL;
      bH.h[0] = *(const v8bf*)&lH[trow * 32 + koff];
      bH.h[1] = *(const v8bf*)&lH[trow * 32 + koff + 16];
      bL.h[0] = *(const v8bf*)&lL[trow * 32 + koff];
      bL.h[1] = *(const v8bf*)&lL[trow * 32 + koff + 16];
      acc[tt] = __builtin_amdgcn_wmma_f32_16x16x32_bf16(false, aH.v, false, bH.v,
                                                        (short)0, acc[tt], false, false);
      acc[tt] = __builtin_amdgcn_wmma_f32_16x16x32_bf16(false, aH.v, false, bL.v,
                                                        (short)0, acc[tt], false, false);
      acc[tt] = __builtin_amdgcn_wmma_f32_16x16x32_bf16(false, aL.v, false, bH.v,
                                                        (short)0, acc[tt], false, false);
    }
    __syncthreads();   // all reads of buf[cur] done before it is refilled
  }

  // ---- rowmax: max over t for each s of this strip ----
  float rmax[8];
#pragma unroll
  for (int r = 0; r < 8; ++r) {
    float m = acc[0][r];
#pragma unroll
    for (int tt = 1; tt < 12; ++tt) m = fmaxf(m, acc[tt][r]);
#pragma unroll
    for (int mask = 1; mask < 16; mask <<= 1)
      m = fmaxf(m, __shfl_xor(m, mask, 32));
    rmax[r] = m;
  }
  if ((lane & 15) == 0) {
    const int sbase = s0 + ((lane >= 16) ? 8 : 0);
    float* dst = pooled + ((size_t)pair * 2 + 1) * SEQ;
#pragma unroll
    for (int r = 0; r < 8; ++r) dst[sbase + r] = rmax[r];
  }

  // ---- colmax: max over s (this strip) per column t, then across waves ----
#pragma unroll
  for (int tt = 0; tt < 12; ++tt) {
    float c = acc[tt][0];
#pragma unroll
    for (int r = 1; r < 8; ++r) c = fmaxf(c, acc[tt][r]);
    c = fmaxf(c, __shfl_xor(c, 16, 32));
    if (lane < 16) colPart[wave * SEQ + tt * 16 + lane] = c;
  }
  __syncthreads();
  if (tid < SEQ) {
    float m = colPart[tid];
#pragma unroll
    for (int w = 1; w < 12; ++w) m = fmaxf(m, colPart[w * SEQ + tid]);
    pooled[((size_t)pair * 2) * SEQ + tid] = m;
  }
}

// ---------------------------------------------------------------------------
// Kernel 2: fused MLP. One wave per 16-row tile of x = BN1(pooled) [9216,192].
// fc2_w comes in as precomputed bf16 hi/lo planes -> no conversion in jt loop.
// ---------------------------------------------------------------------------
__global__ __launch_bounds__(128)
void mlp_kernel(const float* __restrict__ pooled,
                const __bf16* __restrict__ fc2Hi,  // [2048,192] bf16 hi
                const __bf16* __restrict__ fc2Lo,  // [2048,192] bf16 lo
                const float* __restrict__ fc2_b,
                const float* __restrict__ fc3_w,
                const float* __restrict__ fc3_b,
                const float* __restrict__ bn1_g, const float* __restrict__ bn1_b,
                const float* __restrict__ bn1_m, const float* __restrict__ bn1_v,
                const float* __restrict__ bn2_g, const float* __restrict__ bn2_b,
                const float* __restrict__ bn2_m, const float* __restrict__ bn2_v,
                const float* __restrict__ bn3_g, const float* __restrict__ bn3_b,
                const float* __restrict__ bn3_m, const float* __restrict__ bn3_v,
                float* __restrict__ out) {          // [Q*K]
  const int tid  = threadIdx.x;
  const int lane = tid & 31;
  const int wave = tid >> 5;
  const int rt   = blockIdx.x * 4 + wave;    // 0..575
  const int r0   = rt * 16;

  const float s1 = bn1_g[0] * rsqrtf(bn1_v[0] + EPSV);
  const float b1 = bn1_b[0] - bn1_m[0] * s1;

  // x-tile as WMMA A operand (6 chunks of K=32), BN1 folded; converted once.
  const int arow = r0 + (lane & 15);
  const int kb   = (lane < 16) ? 0 : 8;
  const float* xrow = pooled + (size_t)arow * SEQ;
  BfVec16 aH[6], aL[6];
#pragma unroll
  for (int c = 0; c < 6; ++c) {
    float tv[16];
    const float4* p0 = (const float4*)(xrow + c * 32 + kb);
    const float4* p1 = (const float4*)(xrow + c * 32 + kb + 16);
    float4 f0 = p0[0], f1 = p0[1], f2 = p1[0], f3 = p1[1];
    tv[0]=f0.x; tv[1]=f0.y; tv[2]=f0.z; tv[3]=f0.w;
    tv[4]=f1.x; tv[5]=f1.y; tv[6]=f1.z; tv[7]=f1.w;
    tv[8]=f2.x; tv[9]=f2.y; tv[10]=f2.z; tv[11]=f2.w;
    tv[12]=f3.x; tv[13]=f3.y; tv[14]=f3.z; tv[15]=f3.w;
#pragma unroll
    for (int e = 0; e < 16; ++e) {
      float x = tv[e] * s1 + b1;
      split_bf16(x, aH[c].e[e], aL[c].e[e]);
    }
  }

  const v8f vzero = {0.f, 0.f, 0.f, 0.f, 0.f, 0.f, 0.f, 0.f};
  float partial[8] = {0.f, 0.f, 0.f, 0.f, 0.f, 0.f, 0.f, 0.f};
  const int koff = (lane < 16) ? 0 : 8;

  for (int jt = 0; jt < DFF / 16; ++jt) {
    const int j = jt * 16 + (lane & 15);
    const float s2    = bn2_g[j] * rsqrtf(bn2_v[j] + EPSV);
    const float bias2 = (fc2_b[j] - bn2_m[j]) * s2 + bn2_b[j];
    const float w3v   = fc3_w[j];

    v8f acc = vzero;
#pragma unroll
    for (int c = 0; c < 6; ++c) {
      const __bf16* wH = fc2Hi + (size_t)j * SEQ + c * 32 + koff;
      const __bf16* wL = fc2Lo + (size_t)j * SEQ + c * 32 + koff;
      BfVec16 bH, bL;
      bH.h[0] = *(const v8bf*)wH;
      bH.h[1] = *(const v8bf*)(wH + 16);
      bL.h[0] = *(const v8bf*)wL;
      bL.h[1] = *(const v8bf*)(wL + 16);
      acc = __builtin_amdgcn_wmma_f32_16x16x32_bf16(false, aH[c].v, false, bH.v,
                                                    (short)0, acc, false, false);
      acc = __builtin_amdgcn_wmma_f32_16x16x32_bf16(false, aH[c].v, false, bL.v,
                                                    (short)0, acc, false, false);
      acc = __builtin_amdgcn_wmma_f32_16x16x32_bf16(false, aL[c].v, false, bH.v,
                                                    (short)0, acc, false, false);
    }
#pragma unroll
    for (int rr = 0; rr < 8; ++rr) {
      float y = acc[rr] * s2 + bias2;       // BN2 (fc2 bias folded)
      y = fmaxf(y, 0.f);                    // ReLU
      partial[rr] += y * w3v;               // fc3 dot
    }
  }

  // reduce over the 16 lanes holding the same row (within each wave half)
#pragma unroll
  for (int rr = 0; rr < 8; ++rr) {
    float m = partial[rr];
#pragma unroll
    for (int mask = 1; mask < 16; mask <<= 1) m += __shfl_xor(m, mask, 32);
    partial[rr] = m;
  }

  if ((lane & 15) == 0) {
    const float f3b = fc3_b[0];
    const float s3 = bn3_g[0] * rsqrtf(bn3_v[0] + EPSV);
    const float b3 = bn3_b[0] - bn3_m[0] * s3;
    const int rbase = r0 + ((lane >= 16) ? 8 : 0);   // even
#pragma unroll
    for (int i = 0; i < 4; ++i) {
      float sum = (partial[2 * i] + f3b) + (partial[2 * i + 1] + f3b);
      out[rbase / 2 + i] = sum * s3 + b3;
    }
  }
}

// ---------------------------------------------------------------------------
extern "C" void kernel_launch(void* const* d_in, const int* in_sizes, int n_in,
                              void* d_out, int out_size, void* d_ws, size_t ws_size,
                              hipStream_t stream) {
  const float* memory   = (const float*)d_in[0];
  const float* features = (const float*)d_in[1];
  const float* fc2_w    = (const float*)d_in[2];
  const float* fc2_b    = (const float*)d_in[3];
  const float* fc3_w    = (const float*)d_in[4];
  const float* fc3_b    = (const float*)d_in[5];
  const float* bn1_g = (const float*)d_in[6];
  const float* bn1_b = (const float*)d_in[7];
  const float* bn1_m = (const float*)d_in[8];
  const float* bn1_v = (const float*)d_in[9];
  const float* bn2_g = (const float*)d_in[10];
  const float* bn2_b = (const float*)d_in[11];
  const float* bn2_m = (const float*)d_in[12];
  const float* bn2_v = (const float*)d_in[13];
  const float* bn3_g = (const float*)d_in[14];
  const float* bn3_b = (const float*)d_in[15];
  const float* bn3_m = (const float*)d_in[16];
  const float* bn3_v = (const float*)d_in[17];

  // workspace carve-up
  char* ws = (char*)d_ws;
  size_t o = 0;
  float*  pooled = (float*)(ws + o);  o += (size_t)2 * QN * KN * SEQ * sizeof(float);
  __bf16* memHi  = (__bf16*)(ws + o); o += (size_t)QN * SEQ * DD * sizeof(__bf16);
  __bf16* memLo  = (__bf16*)(ws + o); o += (size_t)QN * SEQ * DD * sizeof(__bf16);
  __bf16* featHi = (__bf16*)(ws + o); o += (size_t)KN * SEQ * DD * sizeof(__bf16);
  __bf16* featLo = (__bf16*)(ws + o); o += (size_t)KN * SEQ * DD * sizeof(__bf16);
  __bf16* fc2Hi  = (__bf16*)(ws + o); o += (size_t)DFF * SEQ * sizeof(__bf16);
  __bf16* fc2Lo  = (__bf16*)(ws + o); o += (size_t)DFF * SEQ * sizeof(__bf16);

  const int n8m = QN * SEQ * DD / 8;
  const int n8f = KN * SEQ * DD / 8;
  const int n8w = DFF * SEQ / 8;
  convert_kernel<<<(n8m + 255) / 256, 256, 0, stream>>>(memory,   memHi,  memLo,  n8m);
  convert_kernel<<<(n8f + 255) / 256, 256, 0, stream>>>(features, featHi, featLo, n8f);
  convert_kernel<<<(n8w + 255) / 256, 256, 0, stream>>>(fc2_w,    fc2Hi,  fc2Lo,  n8w);

  score_pool_kernel<<<QN * KN, 384, 0, stream>>>(memHi, memLo, featHi, featLo, pooled);

  mlp_kernel<<<(QN * KN * 2 / 16) / 4, 128, 0, stream>>>(
      pooled, fc2Hi, fc2Lo, fc2_b, fc3_w, fc3_b,
      bn1_g, bn1_b, bn1_m, bn1_v,
      bn2_g, bn2_b, bn2_m, bn2_v,
      bn3_g, bn3_b, bn3_m, bn3_v,
      (float*)d_out);
}
